// BagleyMoELayer_8761733284181
// MI455X (gfx1250) — matrix operand order repaired and checked by reading, not compiled, not run on hardware
//
#include <hip/hip_runtime.h>
#include <hip/hip_bf16.h>

// Problem constants (from reference): B=2, S=1024 -> T=2048 tokens
#define TOK 2048
#define HD  2048
#define ID  1024
#define NE  8
#define NSH 2
#define AUXC 0.01f

typedef __attribute__((ext_vector_type(16))) __bf16 v16bf;
typedef __attribute__((ext_vector_type(8)))  float  v8f;

union FragAB { v16bf v; unsigned u[8]; };

__device__ __forceinline__ unsigned short f2bf(float f) {
  unsigned u = __builtin_bit_cast(unsigned, f);
  u += 0x7FFFu + ((u >> 16) & 1u);   // round-to-nearest-even
  return (unsigned short)(u >> 16);
}

__device__ __forceinline__ v8f vzero8() {
  v8f z;
#pragma unroll
  for (int i = 0; i < 8; ++i) z[i] = 0.0f;
  return z;
}

// CDNA5 async global->LDS copy, 16 bytes per lane (tracked by ASYNCcnt).
__device__ __forceinline__ void async_ld_b128(unsigned short* lds, const unsigned short* g) {
  unsigned laddr = (unsigned)(unsigned long long)lds;   // LDS aperture: low 32 bits = LDS offset
  asm volatile("global_load_async_to_lds_b128 %0, %1, off"
               :: "v"(laddr), "v"((unsigned long long)g) : "memory");
}
__device__ __forceinline__ void wait_async0() {
  asm volatile("s_wait_asynccnt 0x0" ::: "memory");
}

// ---------------------------------------------------------------- zero init
__global__ void zero_kernel(float* __restrict__ out, int n, int* __restrict__ cnt) {
  int i = blockIdx.x * blockDim.x + threadIdx.x;
  if (i < n) out[i] = 0.0f;
  if (i < NE) cnt[i] = 0;
}

// ---------------------------------------------------------------- router
// Writes probs [T,E]; scatters token ids + normalized top-2 weights into
// per-expert buckets (idxbuf/wgtbuf, slot from atomic counter).
__global__ void router_kernel(const float* __restrict__ X, const float* __restrict__ GW,
                              float* __restrict__ probs, int* __restrict__ idxbuf,
                              float* __restrict__ wgtbuf, int* __restrict__ cnt) {
  int t = blockIdx.x * blockDim.x + threadIdx.x;
  if (t >= TOK) return;
  const float* x = X + (size_t)t * HD;
  float lg[NE];
#pragma unroll
  for (int e = 0; e < NE; ++e) lg[e] = 0.0f;
  for (int h = 0; h < HD; ++h) {
    float xv = x[h];
    const float4* g = (const float4*)(GW + (size_t)h * NE);
    float4 a = g[0], b = g[1];
    lg[0] += xv * a.x; lg[1] += xv * a.y; lg[2] += xv * a.z; lg[3] += xv * a.w;
    lg[4] += xv * b.x; lg[5] += xv * b.y; lg[6] += xv * b.z; lg[7] += xv * b.w;
  }
  float mx = lg[0];
#pragma unroll
  for (int e = 1; e < NE; ++e) mx = fmaxf(mx, lg[e]);
  float pr[NE], sum = 0.0f;
#pragma unroll
  for (int e = 0; e < NE; ++e) { pr[e] = __expf(lg[e] - mx); sum += pr[e]; }
  float inv = 1.0f / sum;
#pragma unroll
  for (int e = 0; e < NE; ++e) { pr[e] *= inv; probs[(size_t)t * NE + e] = pr[e]; }
  // top-2 (first index wins ties, matching lax.top_k)
  int i0 = 0;
#pragma unroll
  for (int e = 1; e < NE; ++e) if (pr[e] > pr[i0]) i0 = e;
  int i1 = (i0 == 0) ? 1 : 0;
#pragma unroll
  for (int e = 0; e < NE; ++e) if (e != i0 && pr[e] > pr[i1]) i1 = e;
  float d = 1.0f / (pr[i0] + pr[i1]);
  int s0 = atomicAdd(&cnt[i0], 1);
  idxbuf[i0 * TOK + s0] = t; wgtbuf[i0 * TOK + s0] = pr[i0] * d;
  int s1 = atomicAdd(&cnt[i1], 1);
  idxbuf[i1 * TOK + s1] = t; wgtbuf[i1 * TOK + s1] = pr[i1] * d;
}

// Deterministic per-expert prob sum: fixed partition + fixed tree order.
__global__ void probsum_kernel(const float* __restrict__ probs, float* __restrict__ probsum) {
  __shared__ float red[256];
  const int e = blockIdx.x, tid = threadIdx.x;
  float s = 0.0f;
  for (int t = tid; t < TOK; t += 256) s += probs[(size_t)t * NE + e];
  red[tid] = s;
  __syncthreads();
  for (int off = 128; off > 0; off >>= 1) {
    if (tid < off) red[tid] += red[tid + off];
    __syncthreads();
  }
  if (tid == 0) probsum[e] = red[0];
}

__global__ void loss_kernel(const int* __restrict__ cnt, const float* __restrict__ probsum,
                            float* __restrict__ loss) {
  if (threadIdx.x == 0 && blockIdx.x == 0) {
    float s = 0.0f;
#pragma unroll
    for (int e = 0; e < NE; ++e) s += (float)cnt[e] * (probsum[e] / (float)TOK);
    loss[0] = (float)NE * s / (float)TOK * AUXC;
  }
}

// ---------------------------------------------------------------- stage 1:
// Hh[slot,i] = silu(Xg@Wg)*(Xg@Wu) over gathered token rows (idx==nullptr ->
// identity mapping, dense).  Block tile 128x64, BK=32, 8 waves, bf16 WMMA.
__global__ __launch_bounds__(256)
void gate_up_kernel(const float* __restrict__ X, const float* __restrict__ Wg,
                    const float* __restrict__ Wu, unsigned short* __restrict__ Hout,
                    const int* __restrict__ idx, const int* __restrict__ cntPtr) {
  const int cnt = cntPtr ? cntPtr[0] : TOK;
  const int m0 = blockIdx.y * 128;
  if (m0 >= cnt || cnt == 0) return;
  __shared__ unsigned short Xs[128][34];   // A tile, row-major bf16
  __shared__ unsigned short Gs[64][34];    // gate W tile, transposed [n][k]
  __shared__ unsigned short Us[64][34];    // up   W tile, transposed [n][k]
  const int tid  = threadIdx.x;
  const int lane = tid & 31, wid = tid >> 5;
  const int n0 = blockIdx.x * 64;
  const int wm = (wid >> 1) * 32, wn = (wid & 1) * 32;

  // token index for each of this thread's 4 A-tile rows (fixed across k)
  int tok[4];
#pragma unroll
  for (int it = 0; it < 4; ++it) {
    int row = it * 32 + (tid >> 3);
    int r = m0 + row; if (r >= cnt) r = cnt - 1;
    tok[it] = idx ? idx[r] : r;
  }

  v8f accG[2][2], accU[2][2];
#pragma unroll
  for (int mt = 0; mt < 2; ++mt)
#pragma unroll
    for (int nt = 0; nt < 2; ++nt) { accG[mt][nt] = vzero8(); accU[mt][nt] = vzero8(); }

  for (int k0 = 0; k0 < HD; k0 += 32) {
#pragma unroll
    for (int it = 0; it < 4; ++it) {
      int idx4 = it * 256 + tid;                 // 1024 float4s
      int row = idx4 >> 3, c4 = (idx4 & 7) << 2;
      const float4 v = *(const float4*)(X + (size_t)tok[it] * HD + k0 + c4);
      *(unsigned*)&Xs[row][c4]     = ((unsigned)f2bf(v.x)) | ((unsigned)f2bf(v.y) << 16);
      *(unsigned*)&Xs[row][c4 + 2] = ((unsigned)f2bf(v.z)) | ((unsigned)f2bf(v.w) << 16);
    }
#pragma unroll
    for (int it = 0; it < 8; ++it) {
      int i2 = it * 256 + tid;                   // 2048 elems
      int k = i2 >> 6, n = i2 & 63;
      size_t go = (size_t)(k0 + k) * ID + n0 + n;
      Gs[n][k] = f2bf(Wg[go]);
      Us[n][k] = f2bf(Wu[go]);
    }
    if (k0 + 32 < HD) {                          // CDNA5 global_prefetch_b8
      __builtin_prefetch(Wg + (size_t)(k0 + 32 + (tid >> 6)) * ID + n0 + (tid & 63), 0, 1);
      __builtin_prefetch(Wu + (size_t)(k0 + 32 + (tid >> 6)) * ID + n0 + (tid & 63), 0, 1);
    }
    __syncthreads();

    const int kb_a = (lane >> 4) << 4;
    const int kb_b = (lane >> 4) << 4;
    FragAB bg[2], bu[2];
#pragma unroll
    for (int nt = 0; nt < 2; ++nt) {
      int col = wn + nt * 16 + (lane & 15);
#pragma unroll
      for (int j = 0; j < 8; ++j) {
        int kk = kb_b + 2 * j;
        bg[nt].u[j] = *(const unsigned*)&Gs[col][kk];
        bu[nt].u[j] = *(const unsigned*)&Us[col][kk];
      }
    }
#pragma unroll
    for (int mt = 0; mt < 2; ++mt) {
      FragAB a;
      int row = wm + mt * 16 + (lane & 15);
#pragma unroll
      for (int j = 0; j < 8; ++j) {
        int kk = kb_a + 2 * j;
        a.u[j] = *(const unsigned*)&Xs[row][kk];
      }
#pragma unroll
      for (int nt = 0; nt < 2; ++nt) {
        accG[mt][nt] = __builtin_amdgcn_wmma_f32_16x16x32_bf16(
            false, a.v, false, bg[nt].v, (short)0, accG[mt][nt], false, false);
        accU[mt][nt] = __builtin_amdgcn_wmma_f32_16x16x32_bf16(
            false, a.v, false, bu[nt].v, (short)0, accU[mt][nt], false, false);
      }
    }
    __syncthreads();
  }
#pragma unroll
  for (int mt = 0; mt < 2; ++mt)
#pragma unroll
    for (int nt = 0; nt < 2; ++nt)
#pragma unroll
      for (int r = 0; r < 8; ++r) {
        float g = accG[mt][nt][r], u = accU[mt][nt][r];
        float h = g / (1.0f + __expf(-g)) * u;
        int m = m0 + wm + mt * 16 + ((lane >> 4) << 3) + r;   // slot row
        int n = n0 + wn + nt * 16 + (lane & 15);
        Hout[(size_t)m * ID + n] = f2bf(h);
      }
}

// ---------------------------------------------------------------- stage 2:
// Out[tok(slot)] += w(slot) * (Hh[slot] @ Wd).  A tile staged with CDNA5
// global_load_async_to_lds_b128 (ASYNCcnt); 80B LDS row pitch keeps 16B
// alignment and is bank-conflict-free for fragment reads (stride 20 dwords).
__global__ __launch_bounds__(256)
void down_kernel(const unsigned short* __restrict__ A,   // Hh, bf16 [slots, ID]
                 const float* __restrict__ Wd,           // [ID, HD]
                 float* __restrict__ Out,                 // [T, HD]
                 const int* __restrict__ idx, const float* __restrict__ wgt,
                 const int* __restrict__ cntPtr, float scale) {
  const int cnt = cntPtr ? cntPtr[0] : TOK;
  const int m0 = blockIdx.y * 128;
  if (m0 >= cnt || cnt == 0) return;
  __shared__ unsigned short As[128][40];   // 80B pitch: 16B-aligned rows
  __shared__ unsigned short Ws_[64][34];
  const int tid  = threadIdx.x;
  const int lane = tid & 31, wid = tid >> 5;
  const int n0 = blockIdx.x * 64;
  const int wm = (wid >> 1) * 32, wn = (wid & 1) * 32;

  v8f acc[2][2];
#pragma unroll
  for (int mt = 0; mt < 2; ++mt)
#pragma unroll
    for (int nt = 0; nt < 2; ++nt) acc[mt][nt] = vzero8();

  for (int k0 = 0; k0 < ID; k0 += 32) {
    // A tile: 128 rows x 64B = 512 b128 async transfers, 2 per thread
#pragma unroll
    for (int it = 0; it < 2; ++it) {
      int i4 = it * 256 + tid;                   // 0..511
      int row = i4 >> 2, seg = (i4 & 3) << 3;    // seg in ushorts (8 = 16B)
      async_ld_b128(&As[row][seg], A + (size_t)(m0 + row) * ID + k0 + seg);
    }
    // B tile: f32 -> bf16 through VALU, transposed store
#pragma unroll
    for (int it = 0; it < 8; ++it) {
      int i2 = it * 256 + tid;
      int k = i2 >> 6, n = i2 & 63;
      Ws_[n][k] = f2bf(Wd[(size_t)(k0 + k) * HD + n0 + n]);
    }
    if (k0 + 32 < ID)
      __builtin_prefetch(Wd + (size_t)(k0 + 32 + (tid >> 6)) * HD + n0 + (tid & 63), 0, 1);
    wait_async0();
    __syncthreads();

    const int kb_a = (lane >> 4) << 4;
    const int kb_b = (lane >> 4) << 4;
    FragAB b[2];
#pragma unroll
    for (int nt = 0; nt < 2; ++nt) {
      int col = wn + nt * 16 + (lane & 15);
#pragma unroll
      for (int j = 0; j < 8; ++j)
        b[nt].u[j] = *(const unsigned*)&Ws_[col][kb_b + 2 * j];
    }
#pragma unroll
    for (int mt = 0; mt < 2; ++mt) {
      FragAB a;
      int row = wm + mt * 16 + (lane & 15);
#pragma unroll
      for (int j = 0; j < 8; ++j) {
        int kk = kb_a + 2 * j;
        a.u[j] = *(const unsigned*)&As[row][kk];
      }
#pragma unroll
      for (int nt = 0; nt < 2; ++nt)
        acc[mt][nt] = __builtin_amdgcn_wmma_f32_16x16x32_bf16(
            false, a.v, false, b[nt].v, (short)0, acc[mt][nt], false, false);
    }
    __syncthreads();
  }
#pragma unroll
  for (int mt = 0; mt < 2; ++mt)
#pragma unroll
    for (int nt = 0; nt < 2; ++nt)
#pragma unroll
      for (int r = 0; r < 8; ++r) {
        int slot = m0 + wm + mt * 16 + ((lane >> 4) << 3) + r;
        if (slot < cnt) {
          int t = idx ? idx[slot] : slot;
          float w = wgt ? wgt[slot] * scale : scale;
          int n = n0 + wn + nt * 16 + (lane & 15);
          Out[(size_t)t * HD + n] += w * acc[mt][nt][r];
        }
      }
}

// ---------------------------------------------------------------- launch
extern "C" void kernel_launch(void* const* d_in, const int* in_sizes, int n_in,
                              void* d_out, int out_size, void* d_ws, size_t ws_size,
                              hipStream_t stream) {
  (void)in_sizes; (void)n_in; (void)out_size; (void)ws_size;
  const float* X   = (const float*)d_in[0];
  const float* GW  = (const float*)d_in[1];
  const float* wg  = (const float*)d_in[2];
  const float* wu  = (const float*)d_in[3];
  const float* wd  = (const float*)d_in[4];
  const float* swg = (const float*)d_in[5];
  const float* swu = (const float*)d_in[6];
  const float* swd = (const float*)d_in[7];
  float* Out = (float*)d_out;

  // ws layout (bytes):
  char* ws = (char*)d_ws;
  float* probs   = (float*)(ws);                  // T*E f32   @ 0       (64KB)
  int*   idxbuf  = (int*)(ws + (64 << 10));       // E*T int   @ 64KB    (64KB)
  float* wgtbuf  = (float*)(ws + (128 << 10));    // E*T f32   @ 128KB   (64KB)
  int*   cnt     = (int*)(ws + (192 << 10));      // E int     @ 192KB
  float* probsum = (float*)(ws + (192 << 10) + 64);
  unsigned short* Hh = (unsigned short*)(ws + (256 << 10));  // T*I bf16 @ 256KB (4MB)

  const int nOut = TOK * HD;
  zero_kernel<<<(nOut + 255) / 256, 256, 0, stream>>>(Out, nOut, cnt);
  router_kernel<<<(TOK + 255) / 256, 256, 0, stream>>>(X, GW, probs, idxbuf, wgtbuf, cnt);
  probsum_kernel<<<NE, 256, 0, stream>>>(probs, probsum);
  loss_kernel<<<1, 64, 0, stream>>>(cnt, probsum, Out + nOut);

  dim3 gridGU(ID / 64, TOK / 128);
  dim3 gridD(HD / 64, TOK / 128);
  // routed experts: gathered top-2 buckets (blocks beyond bucket count exit)
  for (int e = 0; e < NE; ++e) {
    gate_up_kernel<<<gridGU, 256, 0, stream>>>(
        X, wg + (size_t)e * HD * ID, wu + (size_t)e * HD * ID, Hh,
        idxbuf + e * TOK, cnt + e);
    down_kernel<<<gridD, 256, 0, stream>>>(
        Hh, wd + (size_t)e * ID * HD, Out,
        idxbuf + e * TOK, wgtbuf + e * TOK, cnt + e, 1.0f);
  }
  // shared experts: dense identity mapping
  for (int s = 0; s < NSH; ++s) {
    gate_up_kernel<<<gridGU, 256, 0, stream>>>(
        X, swg + (size_t)s * HD * ID, swu + (size_t)s * HD * ID, Hh,
        nullptr, nullptr);
    down_kernel<<<gridD, 256, 0, stream>>>(
        Hh, swd + (size_t)s * ID * HD, Out,
        nullptr, nullptr, nullptr, 1.0f / NSH);
  }
}
